// System_4440996184912
// MI455X (gfx1250) — compile-verified, matrix operand order checked by reading, not run on hardware
//
#include <hip/hip_runtime.h>

// Thermostat-controller scan on MI455X (gfx1250, wave32).
// Per wave: 16 batch rows ride the WMMA N dimension; W1ᵀ/W2ᵀ are constant
// A-operands; layer-1 f32 C-layout converts per-lane (no shuffles) into the
// layer-2 f16 B-layout. 2 independent tiles per wave for ILP across the
// sequential recurrence. Activations use gfx1250 V_TANH_F32 (TRANS32,
// co-executes with WMMA). dt loads are software-pipelined one step ahead so
// VMEM latency is off the recurrence critical path.

#define HZ      2048
#define OUT_T   (HZ - 1)      // 2047 output timesteps
#define NSTEPS  (HZ - 2)      // 2046 scan steps
#define BATCH   4096
#define TILES   2             // independent recurrence tiles per wave

typedef __attribute__((ext_vector_type(16))) _Float16 v16h;
typedef __attribute__((ext_vector_type(8)))  float    v8f;

#define LOG2E 1.4426950408889634f

__device__ __forceinline__ float fast_tanh(float v) {
#if __has_builtin(__builtin_amdgcn_tanhf)
    return __builtin_amdgcn_tanhf(v);            // v_tanh_f32 (gfx1250)
#else
    // branchless: tanh(v) = 1 - 2/(exp(2v)+1); exp2 saturation gives ±1 tails
    float e = __builtin_amdgcn_exp2f(2.0f * LOG2E * v);
    return 1.0f - 2.0f * __builtin_amdgcn_rcpf(e + 1.0f);
#endif
}

__device__ __forceinline__ float fast_sigmoid(float v) {
    // 1/(1+exp(-v)), branchless via native exp2 + rcp
    float e = __builtin_amdgcn_exp2f(-LOG2E * v);
    return __builtin_amdgcn_rcpf(1.0f + e);
}

__global__ __launch_bounds__(256) void thermostat_scan_wmma(
    const float* __restrict__ d,    // (4096, 2048)
    const float* __restrict__ W1,   // (2, 64)
    const float* __restrict__ b1,   // (64)
    const float* __restrict__ W2,   // (64, 2)
    const float* __restrict__ b2,   // (2)
    float* __restrict__ xs,         // (4096, 2047)
    float* __restrict__ us)         // (4096, 2047)
{
    const int  lane = threadIdx.x & 31;
    const int  wid  = (blockIdx.x * blockDim.x + threadIdx.x) >> 5;
    const int  sub  = lane & 15;         // batch sub-index / matrix row index
    const bool lo   = lane < 16;

    // ---- constant A operands (live across all steps) ----
    // Layer 1: A1[g] = rows hid=16g..16g+15 of [W1[0]; W1[1]; b1] over K=0..2
    // f16 A 16x32 layout: lanes 0-15 hold K=0..7 (V0-3) & 16..23 (V4-7);
    // lanes 16-31 hold K=8..15 & 24..31  -> upper lanes are all-zero here.
    v16h A1[4];
    #pragma unroll
    for (int g = 0; g < 4; ++g) {
        #pragma unroll
        for (int i = 0; i < 16; ++i) A1[g][i] = (_Float16)0.0f;
        if (lo) {
            const int h = g * 16 + sub;
            A1[g][0] = (_Float16)W1[h];        // W1[0][h] * x
            A1[g][1] = (_Float16)W1[64 + h];   // W1[1][h] * dt
            A1[g][2] = (_Float16)b1[h];        // b1[h]    * 1
        }
    }
    // Layer 2: A2 = W2ᵀ padded to 16x64; row m(=sub) valid for m<2.
    v16h A2a, A2b;
    #pragma unroll
    for (int i = 0; i < 16; ++i) { A2a[i] = (_Float16)0.0f; A2b[i] = (_Float16)0.0f; }
    if (sub < 2) {
        const int m  = sub;
        const int k0 = lo ? 0 : 8;     // K block in V0-3
        const int k1 = lo ? 16 : 24;   // K block in V4-7
        #pragma unroll
        for (int j = 0; j < 8; ++j) {
            A2a[j]     = (_Float16)W2[(k0 + j) * 2 + m];
            A2a[8 + j] = (_Float16)W2[(k1 + j) * 2 + m];
            A2b[j]     = (_Float16)W2[(32 + k0 + j) * 2 + m];
            A2b[8 + j] = (_Float16)W2[(32 + k1 + j) * 2 + m];
        }
    }
    // Layer-2 C init: b2 broadcast into rows M=0,1 (lanes 0-15, V0/V1).
    v8f c2init;
    #pragma unroll
    for (int i = 0; i < 8; ++i) c2init[i] = 0.0f;
    if (lo) { c2init[0] = b2[0]; c2init[1] = b2[1]; }
    v8f zero8;
    #pragma unroll
    for (int i = 0; i < 8; ++i) zero8[i] = 0.0f;

    // ---- recurrence state: x0 = d[:,0]; emit t=0 outputs ----
    int   row[TILES];
    float x[TILES];
    float dtv[TILES];                // software-pipelined dt (one step ahead)
    v16h  B1[TILES];                 // loop-invariant parts built once
    #pragma unroll
    for (int s = 0; s < TILES; ++s) {
        row[s] = wid * (16 * TILES) + s * 16 + sub;   // lanes 16-31 duplicate
        x[s]   = d[(size_t)row[s] * HZ];
        dtv[s] = d[(size_t)row[s] * HZ + 1];          // dt for step t=0
        #pragma unroll
        for (int i = 0; i < 16; ++i) B1[s][i] = (_Float16)0.0f;
        B1[s][2] = lo ? (_Float16)1.0f : (_Float16)0.0f;   // bias row K=2
        if (lo) {
            __builtin_nontemporal_store(x[s], &xs[(size_t)row[s] * OUT_T]);
            __builtin_nontemporal_store(0.0f, &us[(size_t)row[s] * OUT_T]);
        }
    }

    const float tau_inv = 1.0f / 0.06f;

    for (int t = 0; t < NSTEPS; ++t) {
        #pragma unroll
        for (int s = 0; s < TILES; ++s) {
            const float dt = dtv[s];
            // Prefetch next step's dt; t+2 <= 2047 always, so no tail guard.
            dtv[s] = d[(size_t)row[s] * HZ + (t + 2)];

            // B1 (32x16 f16): K=0:x, K=1:dt updated per step; rest invariant.
            B1[s][0] = lo ? (_Float16)x[s] : (_Float16)0.0f;
            B1[s][1] = lo ? (_Float16)dt   : (_Float16)0.0f;

            // Layer 1: H[64x16] pre-activation, 4x WMMA (hid groups in M)
            v8f H0 = __builtin_amdgcn_wmma_f32_16x16x32_f16(false, A1[0], false, B1[s], (short)0, zero8, false, false);
            v8f H1 = __builtin_amdgcn_wmma_f32_16x16x32_f16(false, A1[1], false, B1[s], (short)0, zero8, false, false);
            v8f H2 = __builtin_amdgcn_wmma_f32_16x16x32_f16(false, A1[2], false, B1[s], (short)0, zero8, false, false);
            v8f H3 = __builtin_amdgcn_wmma_f32_16x16x32_f16(false, A1[3], false, B1[s], (short)0, zero8, false, false);

            // tanh + per-lane repack: f32 C-layout -> f16 B-layout (64x16).
            v16h B2a, B2b;
            #pragma unroll
            for (int i = 0; i < 8; ++i) {
                B2a[i]     = (_Float16)fast_tanh(H0[i]);   // K = 0..15 region
                B2a[8 + i] = (_Float16)fast_tanh(H1[i]);   // K = 16..31 region
                B2b[i]     = (_Float16)fast_tanh(H2[i]);   // K = 32..47 region
                B2b[8 + i] = (_Float16)fast_tanh(H3[i]);   // K = 48..63 region
            }

            // Layer 2: zᵀ = W2ᵀ·H + b2, chained K=32 WMMAs.
            v8f Z = __builtin_amdgcn_wmma_f32_16x16x32_f16(false, A2a, false, B2a, (short)0, c2init, false, false);
            Z     = __builtin_amdgcn_wmma_f32_16x16x32_f16(false, A2b, false, B2b, (short)0, Z,      false, false);

            // lane n (n<16): V0 = z0[n], V1 = z1[n]
            const float z0 = Z[0];
            const float z1 = Z[1];
            const float ub = fast_sigmoid(z1 * tau_inv);
            x[s] = 0.99f * x[s] + 0.4f * ub - dt;

            if (lo) {
                __builtin_nontemporal_store(x[s], &xs[(size_t)row[s] * OUT_T + t + 1]);
                __builtin_nontemporal_store(z0,   &us[(size_t)row[s] * OUT_T + t + 1]);
            }
        }
    }
}

extern "C" void kernel_launch(void* const* d_in, const int* in_sizes, int n_in,
                              void* d_out, int out_size, void* d_ws, size_t ws_size,
                              hipStream_t stream) {
    const float* d  = (const float*)d_in[0];
    const float* W1 = (const float*)d_in[1];
    const float* b1 = (const float*)d_in[2];
    const float* W2 = (const float*)d_in[3];
    const float* b2 = (const float*)d_in[4];

    float* xs = (float*)d_out;
    float* us = xs + (size_t)BATCH * OUT_T;   // tuple outputs concatenated flat

    const int waves   = BATCH / (16 * TILES); // 128 waves, 32 rows each
    const int threads = 256;                  // 8 waves per workgroup
    const int blocks  = (waves * 32) / threads;
    thermostat_scan_wmma<<<blocks, threads, 0, stream>>>(d, W1, b1, W2, b2, xs, us);
}
